// SSR_36180804501806
// MI455X (gfx1250) — matrix operand order repaired and checked by reading, not compiled, not run on hardware
//
#include <hip/hip_runtime.h>

// Fused 5x5-window cross-attention (B=4, C=31, H=W=160, Ci=32) for gfx1250.
// Single pass over HBM (bandwidth-bound problem); projections via
// v_wmma_f32_16x16x32_bf16; score/softmax/aggregate in VALU from LDS.
// Pixel-major LDS, f32 kf/q (no pack on hot path), bf16 source halo packed
// two-at-a-time with v_perm_b32. Phase 4 uses paired accumulators + fmaf to
// enable v_pk_fma_f32 / VOPD dual-issue, exp2-domain softmax.

typedef __bf16 v16bf __attribute__((ext_vector_type(16)));
typedef float  v8f   __attribute__((ext_vector_type(8)));

#define WIN      5
#define PADR     2
#define C_IN     31
#define C_INT    32
#define HDIM     160
#define WDIM     160
#define HW       (HDIM * WDIM)
#define TX       16
#define TY       8
#define TILE_PX  (TX * TY)          // 128
#define HX       (TX + 2 * PADR)    // 20
#define HY       (TY + 2 * PADR)    // 12
#define HALO_PX  (HX * HY)          // 240
#define NTHREADS 128
#define NWAVES   4
#define KTILES   (HALO_PX / 16)     // 15
#define QTILES   (TILE_PX / 16)     // 8

// Pack two f32 into a bf16 pair (truncating): D = {hi16(f1), hi16(f0)}.
__device__ __forceinline__ unsigned pk_bf16(float f0, float f1) {
    return __builtin_amdgcn_perm(__float_as_uint(f1), __float_as_uint(f0),
                                 0x07060302u);
}
__device__ __forceinline__ float bflo2f(unsigned u) {   // low bf16 of dword
    union { unsigned u; float f; } v; v.u = u << 16;
    return v.f;
}
__device__ __forceinline__ float bfhi2f(unsigned u) {   // high bf16 of dword
    union { unsigned u; float f; } v; v.u = u & 0xFFFF0000u;
    return v.f;
}

__global__ __launch_bounds__(NTHREADS, 1)
void SSR_win_attn_kernel(const float* __restrict__ anchor,
                         const float* __restrict__ source,
                         const float* __restrict__ Wq,
                         const float* __restrict__ bq,
                         const float* __restrict__ Wk,
                         const float* __restrict__ bk,
                         float* __restrict__ out) {
    // Pixel-major LDS staging. Total 62464 B.
    __shared__ unsigned s_src[HALO_PX * (C_INT / 2)]; // bf16 pairs, ch31 = 0
    __shared__ float    s_kf [HALO_PX * C_INT];       // kf = Wk*src + bk (f32)
    __shared__ float    s_q  [TILE_PX * C_INT];       // q  = Wq*anchor + bq (f32)

    const int tid  = threadIdx.x;
    const int lane = tid & 31;
    const int wid  = __builtin_amdgcn_readfirstlane(tid >> 5);  // SGPR: scalar loops
    const int mlo  = lane & 15;

    // grid: 4 batches * 20 y-tiles * 10 x-tiles = 800 workgroups
    const int blk = blockIdx.x;
    const int b   = blk / ((HDIM / TY) * (WDIM / TX));
    const int rem = blk % ((HDIM / TY) * (WDIM / TX));
    const int ty  = rem / (WDIM / TX);
    const int tx  = rem % (WDIM / TX);
    const int gx0 = tx * TX;
    const int gy0 = ty * TY;

    // ---- Phase 1: source halo -> LDS as bf16 pairs (zero-padded) ----
    for (int i = tid; i < (C_INT / 2) * HALO_PX; i += NTHREADS) {
        const int cp = i / HALO_PX;            // channel pair 0..15
        const int hp = i % HALO_PX;
        const int gx = gx0 + (hp % HX) - PADR;
        const int gy = gy0 + (hp / HX) - PADR;
        float v0 = 0.0f, v1 = 0.0f;
        if ((unsigned)gx < (unsigned)WDIM && (unsigned)gy < (unsigned)HDIM) {
            const float* sp = &source[((b * C_IN + 2 * cp) * HDIM + gy) * WDIM + gx];
            v0 = sp[0];
            if (2 * cp + 1 < C_IN) v1 = sp[HW];
        }
        s_src[hp * (C_INT / 2) + cp] = pk_bf16(v0, v1);
    }

    // ---- Build A fragments (Wk / Wq, Ci=32 -> two 16-row M-tiles, K=32) ----
    // A order used here (contiguous K, B matches): lanes 0-15 row=lane, K in {0..15};
    // lanes 16-31 row=lane-16, K in {16..31}; dword j: K=2*kb+2j.
    const int kb = (lane < 16) ? 0 : 8;
    unsigned aq0[8], aq1[8], ak0[8], ak1[8];
    #pragma unroll
    for (int j = 0; j < 8; ++j) {
        const int k0 = 2 * kb + 2 * j;
        const int k1 = k0 + 1;
        const float q00 = (k0 < C_IN) ? Wq[mlo * C_IN + k0] : 0.f;
        const float q01 = (k1 < C_IN) ? Wq[mlo * C_IN + k1] : 0.f;
        const float q10 = (k0 < C_IN) ? Wq[(mlo + 16) * C_IN + k0] : 0.f;
        const float q11 = (k1 < C_IN) ? Wq[(mlo + 16) * C_IN + k1] : 0.f;
        const float k00 = (k0 < C_IN) ? Wk[mlo * C_IN + k0] : 0.f;
        const float k01 = (k1 < C_IN) ? Wk[mlo * C_IN + k1] : 0.f;
        const float k10 = (k0 < C_IN) ? Wk[(mlo + 16) * C_IN + k0] : 0.f;
        const float k11 = (k1 < C_IN) ? Wk[(mlo + 16) * C_IN + k1] : 0.f;
        aq0[j] = pk_bf16(q00, q01);  aq1[j] = pk_bf16(q10, q11);
        ak0[j] = pk_bf16(k00, k01);  ak1[j] = pk_bf16(k10, k11);
    }
    v16bf aQ0, aQ1, aK0, aK1;
    __builtin_memcpy(&aQ0, aq0, 32); __builtin_memcpy(&aQ1, aq1, 32);
    __builtin_memcpy(&aK0, ak0, 32); __builtin_memcpy(&aK1, ak1, 32);

    // Per-lane bias slices for the D-store (rows rbase..rbase+7 / +16).
    const int rbase = (lane < 16) ? 0 : 8;
    const int kcol  = (lane < 16) ? 0 : 16;       // B-fragment K offset
    float bk0[8], bk1[8], bq0[8], bq1[8];
    #pragma unroll
    for (int r = 0; r < 8; ++r) {
        bk0[r] = bk[rbase + r];      bk1[r] = bk[16 + rbase + r];
        bq0[r] = bq[rbase + r];      bq1[r] = bq[16 + rbase + r];
    }

    __syncthreads();

    // ---- Phase 2: kf projection over halo (WMMA), 15 pixel-tiles / 4 waves ----
    for (int t = wid; t < KTILES; t += NWAVES) {   // scalar loop: EXEC untouched
        const int px = t * 16 + mlo;               // halo pixel (column of B/D)
        v16bf bfrag;                               // 32B contiguous -> 2x ds_load_b128
        __builtin_memcpy(&bfrag, &s_src[px * (C_INT / 2) + kcol / 2], 32);
        v8f c0 = {}, c1 = {};
        c0 = __builtin_amdgcn_wmma_f32_16x16x32_bf16(false, aK0, false, bfrag,
                                                     (short)0, c0, false, false);
        c1 = __builtin_amdgcn_wmma_f32_16x16x32_bf16(false, aK1, false, bfrag,
                                                     (short)0, c1, false, false);
        float* kp = &s_kf[px * C_INT];
        #pragma unroll
        for (int r = 0; r < 8; ++r) {              // contiguous 8-float runs
            kp[rbase + r]      = c0[r] + bk0[r];
            kp[16 + rbase + r] = c1[r] + bk1[r];
        }
    }

    // ---- Phase 3: q projection over tile (WMMA), 8 pixel-tiles / 4 waves ----
    for (int t = wid; t < QTILES; t += NWAVES) {
        const int p  = t * 16 + mlo;               // pixel within 16x8 tile
        const int gx = gx0 + (p % TX);
        const int gy = gy0 + (p / TX);
        const float* ap = &anchor[((b * C_IN + kcol) * HDIM + gy) * WDIM + gx];
        unsigned bu[8];
        #pragma unroll
        for (int j = 0; j < 8; ++j) {              // B elem e: K = kcol + e
            const float f0 = ap[(2 * j) * HW];
            const float f1 = (kcol + 2 * j + 1 < C_IN) ? ap[(2 * j + 1) * HW] : 0.f;
            bu[j] = pk_bf16(f0, f1);
        }
        v16bf bfrag;
        __builtin_memcpy(&bfrag, bu, 32);
        v8f c0 = {}, c1 = {};
        c0 = __builtin_amdgcn_wmma_f32_16x16x32_bf16(false, aQ0, false, bfrag,
                                                     (short)0, c0, false, false);
        c1 = __builtin_amdgcn_wmma_f32_16x16x32_bf16(false, aQ1, false, bfrag,
                                                     (short)0, c1, false, false);
        float* qp = &s_q[p * C_INT];
        #pragma unroll
        for (int r = 0; r < 8; ++r) {
            qp[rbase + r]      = c0[r] + bq0[r];
            qp[16 + rbase + r] = c1[r] + bq1[r];
        }
    }

    __syncthreads();

    // ---- Phase 4: per-pixel score -> softmax(25) -> value aggregation ----
    const int p  = tid;            // 128 threads == 128 tile pixels
    const int px = p % TX;
    const int py = p / TX;

    // Pre-scale q by (Ci^-0.5 * log2(e)) so scores live in exp2 domain and
    // no per-window scaling is needed.
    const float qscale = 0.17677669529663687f * 1.44269504088896340f;
    float qv[C_INT];
    {
        const float4* qp = (const float4*)&s_q[p * C_INT];   // 128B-aligned
        #pragma unroll
        for (int i = 0; i < C_INT / 4; ++i) {
            const float4 t4 = qp[i];
            qv[4 * i + 0] = t4.x * qscale; qv[4 * i + 1] = t4.y * qscale;
            qv[4 * i + 2] = t4.z * qscale; qv[4 * i + 3] = t4.w * qscale;
        }
    }

    float sc[WIN * WIN];
    float smax = -3.0e38f;
    #pragma unroll
    for (int k = 0; k < WIN * WIN; ++k) {
        const int hp = (py + k / WIN) * HX + (px + k % WIN);  // halo-relative
        const float4* kp = (const float4*)&s_kf[hp * C_INT];
        // Independent partial sums -> v_pk_fma_f32 / VOPD candidates.
        float s0 = 0.f, s1 = 0.f, s2 = 0.f, s3 = 0.f;
        #pragma unroll
        for (int i = 0; i < C_INT / 4; ++i) {
            const float4 t4 = kp[i];
            s0 = fmaf(qv[4 * i + 0], t4.x, s0);
            s1 = fmaf(qv[4 * i + 1], t4.y, s1);
            s2 = fmaf(qv[4 * i + 2], t4.z, s2);
            s3 = fmaf(qv[4 * i + 3], t4.w, s3);
        }
        const float s = (s0 + s2) + (s1 + s3);
        sc[k] = s;
        smax = fmaxf(smax, s);
    }
    float ssum = 0.0f;
    #pragma unroll
    for (int k = 0; k < WIN * WIN; ++k) {
        sc[k] = __builtin_amdgcn_exp2f(sc[k] - smax);  // single v_exp_f32
        ssum += sc[k];
    }
    const float inv = 1.0f / ssum;         // folded into final scale

    // Paired accumulators (adjacent channels) -> packed/dual-issued FMAs.
    float acc[C_INT];
    #pragma unroll
    for (int c = 0; c < C_INT; ++c) acc[c] = 0.0f;
    #pragma unroll
    for (int k = 0; k < WIN * WIN; ++k) {
        const float w = sc[k];             // un-normalized; *inv applied at end
        const int hp = (py + k / WIN) * HX + (px + k % WIN);
        const uint4* vp = (const uint4*)&s_src[hp * (C_INT / 2)]; // 64B per pixel
        #pragma unroll
        for (int j = 0; j < 4; ++j) {      // 4x uint4 = 32 bf16 channels
            const uint4 u4 = vp[j];
            const int c0 = j * 8;
            acc[c0 + 0] = fmaf(w, bflo2f(u4.x), acc[c0 + 0]);
            acc[c0 + 1] = fmaf(w, bfhi2f(u4.x), acc[c0 + 1]);
            acc[c0 + 2] = fmaf(w, bflo2f(u4.y), acc[c0 + 2]);
            acc[c0 + 3] = fmaf(w, bfhi2f(u4.y), acc[c0 + 3]);
            acc[c0 + 4] = fmaf(w, bflo2f(u4.z), acc[c0 + 4]);
            acc[c0 + 5] = fmaf(w, bfhi2f(u4.z), acc[c0 + 5]);
            acc[c0 + 6] = fmaf(w, bflo2f(u4.w), acc[c0 + 6]);
            if (c0 + 7 < C_IN)             // skip padded channel 31
                acc[c0 + 7] = fmaf(w, bfhi2f(u4.w), acc[c0 + 7]);
        }
    }

    const int gx = gx0 + px;
    const int gy = gy0 + py;
    #pragma unroll
    for (int c = 0; c < C_IN; ++c)
        out[((b * C_IN + c) * HDIM + gy) * WDIM + gx] = acc[c] * inv;
}

extern "C" void kernel_launch(void* const* d_in, const int* in_sizes, int n_in,
                              void* d_out, int out_size, void* d_ws, size_t ws_size,
                              hipStream_t stream) {
    (void)in_sizes; (void)n_in; (void)out_size; (void)d_ws; (void)ws_size;
    const float* anchor = (const float*)d_in[0];
    const float* source = (const float*)d_in[1];
    const float* Wq     = (const float*)d_in[2];
    const float* bq     = (const float*)d_in[3];
    const float* Wk     = (const float*)d_in[4];
    const float* bk     = (const float*)d_in[5];
    float*       out    = (float*)d_out;

    const int nblocks = 4 * (HDIM / TY) * (WDIM / TX);  // 4 * 20 * 10 = 800
    SSR_win_attn_kernel<<<dim3(nblocks), dim3(NTHREADS), 0, stream>>>(
        anchor, source, Wq, bq, Wk, bk, out);
}
